// MultiHeadAttention_70085276336758
// MI455X (gfx1250) — matrix operand order repaired and checked by reading, not compiled, not run on hardware
//
#include <hip/hip_runtime.h>

typedef __attribute__((ext_vector_type(16))) _Float16 v16h;
typedef __attribute__((ext_vector_type(8)))  float    v8f;
typedef __attribute__((ext_vector_type(4)))  int      v4i;

#define BATCH 4
#define SEQ   2048
#define DIM   1024
#define NH    16
#define HD    64

union H8 { uint4 u4; _Float16 h[8]; };
union H4 { uint2 u2; _Float16 h[4]; };

// ---------------------------------------------------------------------------
// CDNA5 async global->LDS copy (ASYNCcnt-tracked, no VGPR staging), guarded.
// Builtin signature (probed via diagnostic): (v4i AS1*, v4i AS3*, imm, imm).
// ---------------------------------------------------------------------------
#if __has_builtin(__builtin_amdgcn_global_load_async_to_lds_b128)
#define HAS_ASYNC_LDS 1
__device__ __forceinline__ void async_copy16(const void* gsrc, void* ldst) {
  __builtin_amdgcn_global_load_async_to_lds_b128(
      (__attribute__((address_space(1))) v4i*)gsrc,
      (__attribute__((address_space(3))) v4i*)ldst, 0, 0);
}
__device__ __forceinline__ void async_wait0() {
#if __has_builtin(__builtin_amdgcn_s_wait_asynccnt)
  __builtin_amdgcn_s_wait_asynccnt(0);
#else
  asm volatile("s_wait_asynccnt 0" ::: "memory");
#endif
}
#else
#define HAS_ASYNC_LDS 0
__device__ __forceinline__ void async_copy16(const void* gsrc, void* ldst) {
  *(uint4*)ldst = *(const uint4*)gsrc;
}
__device__ __forceinline__ void async_wait0() {}
#endif

// Build a 16-half WMMA fragment from two 16-byte chunks.
// A-matrix use: lo = K[kb..kb+7], hi = K[kb+8..kb+15]   (kb  = lane<16 ? 0 : 16)
// B-matrix use: lo = K[kb2..kb2+7], hi = K[kb2+8..kb2+15] (kb2 = lane<16 ? 0 : 16)
__device__ __forceinline__ v16h frag_ld(const _Float16* lo, const _Float16* hi) {
  H8 a, b;
  a.u4 = *(const uint4*)lo;
  b.u4 = *(const uint4*)hi;
  v16h f;
#pragma unroll
  for (int i = 0; i < 8; ++i) { f[i] = a.h[i]; f[8 + i] = b.h[i]; }
  return f;
}

__device__ __forceinline__ v8f wmma_f16(v16h a, v16h b, v8f c) {
  return __builtin_amdgcn_wmma_f32_16x16x32_f16(false, a, false, b, (short)0, c,
                                                false, false);
}

__device__ __forceinline__ v8f vzero8() {
  v8f z;
#pragma unroll
  for (int i = 0; i < 8; ++i) z[i] = 0.0f;
  return z;
}

// reductions across a 16-lane half-wave (masks <=8 never cross the group)
__device__ __forceinline__ float rmax16(float v) {
#pragma unroll
  for (int m = 8; m >= 1; m >>= 1) v = fmaxf(v, __shfl_xor(v, m, 32));
  return v;
}
__device__ __forceinline__ float rsum16(float v) {
#pragma unroll
  for (int m = 8; m >= 1; m >>= 1) v += __shfl_xor(v, m, 32);
  return v;
}

// ---------------------------------------------------------------------------
// Kernel 1: fused QKV projection.
// grid.x = 3 * NH * (8192/64); block = 128 (4 waves).
// Block computes a 64(M) x 64(N=one head) tile of Y = X * W[h] in f16.
// Output layout: [B, H, S, HD] (head-contiguous), f16.
// ---------------------------------------------------------------------------
__global__ __launch_bounds__(128)
void qkv_proj_kernel(const float* __restrict__ q, const float* __restrict__ k,
                     const float* __restrict__ v, const float* __restrict__ Wq,
                     const float* __restrict__ Wk, const float* __restrict__ Wv,
                     _Float16* __restrict__ ws) {
  __shared__ __align__(16) _Float16 Al[64 * 32];  // [row][k]
  __shared__ __align__(16) _Float16 Bl[64 * 32];  // [n][k] (transposed)

  const int bx    = blockIdx.x;
  const int which = bx >> 11;      // 0:q 1:k 2:v (2048 blocks each)
  const int rem   = bx & 2047;
  const int h     = rem >> 7;      // head
  const int m0    = (rem & 127) * 64;

  const float* X = (which == 0) ? q : (which == 1) ? k : v;
  const float* W = (which == 0) ? Wq : (which == 1) ? Wk : Wv;
  const float* Wh = W + (size_t)h * DIM * HD;
  _Float16* Y = ws + (size_t)which * ((size_t)BATCH * NH * SEQ * HD);

  const int tid   = threadIdx.x;
  const int wave  = tid >> 5;
  const int lane  = tid & 31;
  const int lhalf = lane & 15;
  const int lg    = lane >> 4;
  const int kbA   = lg ? 16 : 0;
  const int kbB   = lg ? 16 : 0;

  v8f acc[4];
#pragma unroll
  for (int t = 0; t < 4; ++t) acc[t] = vzero8();

  for (int k0 = 0; k0 < DIM; k0 += 32) {
    __syncthreads();
    // A tile: 64 rows x 32 k. 512 float4 chunks; float4 load -> b64 LDS store.
#pragma unroll
    for (int i = 0; i < 4; ++i) {
      int c = tid + i * 128;                // float4 chunk id
      int row = c >> 3, kq = (c & 7) * 4;
      float4 f = *(const float4*)(X + (size_t)(m0 + row) * DIM + (k0 + kq));
      H4 d;
      d.h[0] = (_Float16)f.x; d.h[1] = (_Float16)f.y;
      d.h[2] = (_Float16)f.z; d.h[3] = (_Float16)f.w;
      *(uint2*)&Al[row * 32 + kq] = d.u2;
    }
    // B tile transposed: Bl[n*32+kk] = W[h][(k0+kk)*HD + n].
    // 4 lane-coalesced b32 loads -> one b64 LDS store.
#pragma unroll
    for (int i = 0; i < 4; ++i) {
      int c = tid + i * 128;
      int n = c & 63, kq = (c >> 6) * 4;
      H4 d;
#pragma unroll
      for (int j = 0; j < 4; ++j)
        d.h[j] = (_Float16)Wh[(size_t)(k0 + kq + j) * HD + n];
      *(uint2*)&Bl[n * 32 + kq] = d.u2;
    }
    __syncthreads();

    const _Float16* arow = Al + (wave * 16 + lhalf) * 32;
    v16h af = frag_ld(arow + kbA, arow + kbA + 8);
#pragma unroll
    for (int t = 0; t < 4; ++t) {
      const _Float16* bcol = Bl + (t * 16 + lhalf) * 32 + kbB;
      v16h bf = frag_ld(bcol, bcol + 8);
      acc[t] = wmma_f16(af, bf, acc[t]);
    }
  }

  // store: C/D layout -> [B,H,S,HD] f16
#pragma unroll
  for (int r = 0; r < 8; ++r) {
    int m = r + 8 * lg;
    int rIdx = m0 + wave * 16 + m;       // global row in [B*S)
    int bb = rIdx >> 11;                 // / SEQ
    int s  = rIdx & 2047;
    size_t rowoff = (((size_t)bb * NH + h) * SEQ + s) * HD;
#pragma unroll
    for (int t = 0; t < 4; ++t)
      Y[rowoff + t * 16 + lhalf] = (_Float16)acc[t][r];
  }
}

// ---------------------------------------------------------------------------
// Kernel 2: causal flash attention per (b, h, 64-query block).
// grid.x = B * NH * (SEQ/64); block = 128 (4 waves, 16 query rows each).
// Reads qh/kh/vh f16 [B,H,S,HD]; writes att f16 [B,H,S,HD].
// K tile staged with GLOBAL_LOAD_ASYNC_TO_LDS_B128 (ASYNCcnt path).
// ---------------------------------------------------------------------------
__global__ __launch_bounds__(128)
void attn_kernel(const _Float16* __restrict__ ws, _Float16* __restrict__ att) {
  const size_t per = (size_t)BATCH * NH * SEQ * HD;
  const _Float16* qh = ws;
  const _Float16* kh = ws + per;
  const _Float16* vh = ws + 2 * per;

  __shared__ __align__(16) _Float16 Kl[32 * 64];     // [key][e]
  __shared__ __align__(16) _Float16 Vt[64 * 32];     // [e][key] (transposed)
  __shared__ __align__(16) _Float16 Pl[4][16 * 32];  // per-wave P scratch

  const int bx  = blockIdx.x;
  const int b   = bx >> 9;
  const int rem = bx & 511;
  const int h   = rem >> 5;
  const int q0  = (rem & 31) * 64;

  const int tid   = threadIdx.x;
  const int wave  = tid >> 5;
  const int lane  = tid & 31;
  const int lhalf = lane & 15;
  const int lg    = lane >> 4;
  const int kbA   = lg ? 16 : 0;
  const int kbB   = lg ? 16 : 0;

  const size_t headoff = ((size_t)b * NH + h) * SEQ * HD;

  // Preload Q fragments for this wave's 16 rows, pre-scaled by 1/sqrt(HD)=0.125
  const _Float16* qp = qh + headoff + (size_t)(q0 + wave * 16 + lhalf) * HD;
  v16h qf[2];
#pragma unroll
  for (int ks = 0; ks < 2; ++ks) {
    qf[ks] = frag_ld(qp + ks * 32 + kbA, qp + ks * 32 + kbA + 8);
#pragma unroll
    for (int i = 0; i < 16; ++i) qf[ks][i] = qf[ks][i] * (_Float16)0.125f;
  }

  v8f o[4];
  float mrow[8], lrow[8];
#pragma unroll
  for (int t = 0; t < 4; ++t) o[t] = vzero8();
#pragma unroll
  for (int r = 0; r < 8; ++r) { mrow[r] = -__builtin_inff(); lrow[r] = 0.0f; }

  const int jmax = q0 + 64;
  for (int j0 = 0; j0 < jmax; j0 += 32) {
    __syncthreads();
    // K tile: [key][e] rows are contiguous -> whole 4KB tile is one flat copy.
    // Async global->LDS, 16B per lane per op, 2 ops per lane.
    {
      const _Float16* kt = kh + headoff + (size_t)j0 * HD;
#pragma unroll
      for (int i = 0; i < 2; ++i) {
        int c = tid + i * 128;  // 16B chunk id, 256 total
        async_copy16(kt + c * 8, Kl + c * 8);
      }
    }
    // V tile transposed [e][key]: uint4 global load + 8 b16 LDS stores.
#pragma unroll
    for (int i = 0; i < 2; ++i) {
      int c = tid + i * 128;                 // 8-half chunk id
      int key = c >> 3, e0 = (c & 7) * 8;
      H8 d;
      d.u4 = *(const uint4*)(vh + headoff + (size_t)(j0 + key) * HD + e0);
#pragma unroll
      for (int j = 0; j < 8; ++j) Vt[(e0 + j) * 32 + key] = d.h[j];
    }
    async_wait0();
    __syncthreads();

    // scores: 16 rows x 32 keys as two 16x16 tiles, K-dim = HD (2 wmma each)
    v8f sc[2];
#pragma unroll
    for (int ns = 0; ns < 2; ++ns) {
      v8f c = vzero8();
#pragma unroll
      for (int ks = 0; ks < 2; ++ks) {
        const _Float16* kp = Kl + (ns * 16 + lhalf) * 64 + ks * 32 + kbB;
        v16h bf = frag_ld(kp, kp + 8);
        c = wmma_f16(qf[ks], bf, c);
      }
      sc[ns] = c;
    }

    // causal mask (element (m,n): m = r + 8*lg, n = lhalf in C/D layout)
#pragma unroll
    for (int r = 0; r < 8; ++r) {
      int qg = q0 + wave * 16 + r + 8 * lg;
#pragma unroll
      for (int ns = 0; ns < 2; ++ns) {
        if (j0 + ns * 16 + lhalf > qg) sc[ns][r] = -__builtin_inff();
      }
    }

    // online softmax update + write P~ (f16) to per-wave LDS scratch
    _Float16* pw = Pl[wave];
#pragma unroll
    for (int r = 0; r < 8; ++r) {
      float bm = fmaxf(rmax16(sc[0][r]), rmax16(sc[1][r]));
      float mn = fmaxf(mrow[r], bm);
      float alpha = __expf(mrow[r] - mn);
      mrow[r] = mn;
      float p0 = __expf(sc[0][r] - mn);
      float p1 = __expf(sc[1][r] - mn);
      lrow[r] = lrow[r] * alpha + rsum16(p0) + rsum16(p1);
#pragma unroll
      for (int t = 0; t < 4; ++t) o[t][r] *= alpha;
      int m = r + 8 * lg;
      pw[m * 32 + lhalf]      = (_Float16)p0;
      pw[m * 32 + 16 + lhalf] = (_Float16)p1;
    }
    // DS ops are in-order within a wave: P writes precede P reads below.

    // O += P~ x V  (K-dim = 32 keys, 4 e-subtiles)
    const _Float16* pr = Pl[wave] + lhalf * 32;
    v16h pf = frag_ld(pr + kbA, pr + kbA + 8);
#pragma unroll
    for (int t = 0; t < 4; ++t) {
      const _Float16* vp = Vt + (t * 16 + lhalf) * 32 + kbB;
      v16h bf = frag_ld(vp, vp + 8);
      o[t] = wmma_f16(pf, bf, o[t]);
    }
  }

  // normalize and store att f16 [B,H,S,HD]
#pragma unroll
  for (int r = 0; r < 8; ++r) {
    int m = r + 8 * lg;
    float inv = 1.0f / lrow[r];
    size_t rowoff = headoff + (size_t)(q0 + wave * 16 + m) * HD;
#pragma unroll
    for (int t = 0; t < 4; ++t)
      att[rowoff + t * 16 + lhalf] = (_Float16)(o[t][r] * inv);
  }
}

// ---------------------------------------------------------------------------
// Kernel 3: output projection. att (flat [8192,1024] f16 -- this flat view IS
// the reference's raw reshape of [B,H,S,hd]) x Wo f32 -> d_out f32.
// grid.x = (8192/64) * (1024/64); block = 128.
// A tile staged via async global->LDS (pure f16 copy).
// ---------------------------------------------------------------------------
__global__ __launch_bounds__(128)
void out_proj_kernel(const _Float16* __restrict__ att,
                     const float* __restrict__ Wo, float* __restrict__ out) {
  __shared__ __align__(16) _Float16 Al[64 * 32];
  __shared__ __align__(16) _Float16 Bl[64 * 32];

  const int bx = blockIdx.x;
  const int m0 = (bx >> 4) * 64;
  const int n0 = (bx & 15) * 64;

  const int tid   = threadIdx.x;
  const int wave  = tid >> 5;
  const int lane  = tid & 31;
  const int lhalf = lane & 15;
  const int lg    = lane >> 4;
  const int kbA   = lg ? 16 : 0;
  const int kbB   = lg ? 16 : 0;

  v8f acc[4];
#pragma unroll
  for (int t = 0; t < 4; ++t) acc[t] = vzero8();

  for (int k0 = 0; k0 < DIM; k0 += 32) {
    __syncthreads();
    // A tile: 64 rows x 32 halves; async 16B copies (rows strided in global).
#pragma unroll
    for (int i = 0; i < 2; ++i) {
      int c = tid + i * 128;                 // 16B chunk id, 256 total
      int row = c >> 2, kq = (c & 3) * 8;
      async_copy16(att + (size_t)(m0 + row) * DIM + (k0 + kq),
                   Al + row * 32 + kq);
    }
    // B tile transposed: Bl[n*32+kk] = Wo[(k0+kk)*DIM + n0+n], f32->f16.
#pragma unroll
    for (int i = 0; i < 4; ++i) {
      int c = tid + i * 128;
      int n = c & 63, kq = (c >> 6) * 4;
      H4 d;
#pragma unroll
      for (int j = 0; j < 4; ++j)
        d.h[j] = (_Float16)Wo[(size_t)(k0 + kq + j) * DIM + (n0 + n)];
      *(uint2*)&Bl[n * 32 + kq] = d.u2;
    }
    async_wait0();
    __syncthreads();

    const _Float16* arow = Al + (wave * 16 + lhalf) * 32;
    v16h af = frag_ld(arow + kbA, arow + kbA + 8);
#pragma unroll
    for (int t = 0; t < 4; ++t) {
      const _Float16* bcol = Bl + (t * 16 + lhalf) * 32 + kbB;
      v16h bf = frag_ld(bcol, bcol + 8);
      acc[t] = wmma_f16(af, bf, acc[t]);
    }
  }

#pragma unroll
  for (int r = 0; r < 8; ++r) {
    int m = r + 8 * lg;
    size_t rowoff = (size_t)(m0 + wave * 16 + m) * DIM;
#pragma unroll
    for (int t = 0; t < 4; ++t)
      out[rowoff + n0 + t * 16 + lhalf] = acc[t][r];
  }
}

extern "C" void kernel_launch(void* const* d_in, const int* in_sizes, int n_in,
                              void* d_out, int out_size, void* d_ws,
                              size_t ws_size, hipStream_t stream) {
  const float* q  = (const float*)d_in[0];
  const float* k  = (const float*)d_in[1];
  const float* v  = (const float*)d_in[2];
  const float* Wq = (const float*)d_in[3];
  const float* Wk = (const float*)d_in[4];
  const float* Wv = (const float*)d_in[5];
  const float* Wo = (const float*)d_in[6];

  _Float16* ws = (_Float16*)d_ws;
  const size_t per = (size_t)BATCH * NH * SEQ * HD;  // 8,388,608 halves
  _Float16* att = ws + 3 * per;                      // total ws: 64 MiB

  // 1) fused QKV projections -> qh/kh/vh f16 [B,H,S,HD]
  qkv_proj_kernel<<<3 * NH * ((BATCH * SEQ) / 64), 128, 0, stream>>>(
      q, k, v, Wq, Wk, Wv, ws);
  // 2) causal flash attention -> att f16 [B,H,S,HD]
  attn_kernel<<<BATCH * NH * (SEQ / 64), 128, 0, stream>>>(ws, att);
  // 3) output projection -> d_out f32 [B,S,D]
  out_proj_kernel<<<((BATCH * SEQ) / 64) * (DIM / 64), 128, 0, stream>>>(
      att, Wo, (float*)d_out);
}
